// GCN_16166256902759
// MI455X (gfx1250) — matrix operand order repaired and checked by reading, not compiled, not run on hardware
//
#include <hip/hip_runtime.h>
#include <hip/hip_bf16.h>

typedef __attribute__((ext_vector_type(16))) _Float16 v16h;
typedef __attribute__((ext_vector_type(8)))  _Float16 v8h;
typedef __attribute__((ext_vector_type(8)))  float    v8f;

#define LDF 512   // f32 buffer row stride (max padded dout)
#define LDC 544   // concat f16 row stride (520 -> 544)

// ---------------------------------------------------------------------------
// utility kernels
// ---------------------------------------------------------------------------
__global__ void fill_kernel(float* __restrict__ p, float v, int n) {
  int i = blockIdx.x * blockDim.x + threadIdx.x;
  if (i < n) p[i] = v;
}

__global__ void deg_kernel(const int* __restrict__ dst, float* __restrict__ deg, int E) {
  int e = blockIdx.x * blockDim.x + threadIdx.x;
  if (e < E) atomicAdd(&deg[dst[e]], 1.0f);
}

__global__ void rsqrt_kernel(float* __restrict__ d, int n) {
  int i = blockIdx.x * blockDim.x + threadIdx.x;
  if (i < n) d[i] = rsqrtf(d[i]);
}

__global__ void norm_kernel(const int* __restrict__ ei, const float* __restrict__ dinv,
                            float* __restrict__ nrm, int E, int n) {
  int m = blockIdx.x * blockDim.x + threadIdx.x;
  if (m >= E + n) return;
  int s, d;
  if (m < E) { s = ei[m]; d = ei[E + m]; } else { s = d = m - E; }
  nrm[m] = dinv[s] * dinv[d];
}

// W [din, dout] f32  ->  WT [doutp, Kp] f16 (transposed, zero-padded)
__global__ void wt_kernel(const float* __restrict__ W, _Float16* __restrict__ WT,
                          int din, int dout, int Kp, int total) {
  int idx = blockIdx.x * blockDim.x + threadIdx.x;
  if (idx >= total) return;
  int c = idx / Kp, k = idx - c * Kp;
  WT[idx] = (c < dout && k < din) ? (_Float16)W[(size_t)k * dout + c] : (_Float16)0.0f;
}

// x [n,8] f32 -> Xh [n,32] f16 zero-padded
__global__ void xh_kernel(const float* __restrict__ x, _Float16* __restrict__ Xh, int n) {
  int idx = blockIdx.x * blockDim.x + threadIdx.x;
  if (idx >= n * 32) return;
  int r = idx >> 5, k = idx & 31;
  Xh[idx] = (k < 8) ? (_Float16)x[r * 8 + k] : (_Float16)0.0f;
}

// ---------------------------------------------------------------------------
// WMMA GEMM: C[n, doutp] = A[n, Kp](f16) @ WT[doutp, Kp]^T (f16), f32 accum.
// 256 threads = 8 waves; each wave owns a 16x32 output tile (2 WMMA / K-step,
// A fragment reused). Operands zero-padded -> branch-free inner loop, operand
// traffic served by L2 (entire working set < 192 MB).
// ---------------------------------------------------------------------------
__global__ __launch_bounds__(256) void gemm_wmma_kernel(
    const _Float16* __restrict__ A, const _Float16* __restrict__ WT,
    float* __restrict__ C, int n, int nRowTiles, int Kp, int doutp)
{
  const int wave = threadIdx.x >> 5;
  const int lane = threadIdx.x & 31;
  const int rt   = blockIdx.x * 8 + wave;    // 16-row tile index (wave-uniform)
  if (rt >= nRowTiles) return;

  const int l15  = lane & 15;
  const int col0 = blockIdx.y * 32;
  const int kbA  = (lane < 16) ? 0 : 16;     // A: contiguous K per lane half
  const int kbB  = (lane < 16) ? 0 : 16;     // B: contiguous K per lane half (same as A)

  const _Float16* ap  = A  + (size_t)(rt * 16 + l15) * Kp;
  const _Float16* bp0 = WT + (size_t)(col0 + l15) * Kp;
  const _Float16* bp1 = bp0 + (size_t)16 * Kp;

  v8f acc0 = {}, acc1 = {};
  for (int k0 = 0; k0 < Kp; k0 += 32) {
    union { v16h v; v8h h[2]; } af;
    af.h[0] = *(const v8h*)(ap + k0 + kbA);        // K in {kbA..kbA+7}
    af.h[1] = *(const v8h*)(ap + k0 + kbA + 8);    // K in {kbA+8..kbA+15}
    v16h b0 = *(const v16h*)(bp0 + k0 + kbB);      // K in {kbB..kbB+15}
    v16h b1 = *(const v16h*)(bp1 + k0 + kbB);
    acc0 = __builtin_amdgcn_wmma_f32_16x16x32_f16(false, af.v, false, b0,
                                                  (short)0, acc0, false, false);
    acc1 = __builtin_amdgcn_wmma_f32_16x16x32_f16(false, af.v, false, b1,
                                                  (short)0, acc1, false, false);
  }

  // D layout: VGPR j -> M = j + (lane<16 ? 0 : 8), N = lane%16
  const int mbase = rt * 16 + ((lane < 16) ? 0 : 8);
  const int c0 = col0 + l15;
#pragma unroll
  for (int j = 0; j < 8; ++j) {
    int gr = mbase + j;
    if (gr < n) {
      C[(size_t)gr * LDF + c0]      = acc0[j];
      C[(size_t)gr * LDF + c0 + 16] = acc1[j];
    }
  }
}

// ---------------------------------------------------------------------------
// gather * norm -> scatter-add (f32 atomics into L2-resident agg buffer)
// ---------------------------------------------------------------------------
__global__ void scatter_kernel(const float* __restrict__ H, const int* __restrict__ ei,
                               const float* __restrict__ nrm, float* __restrict__ out,
                               int E, int n, int dout)
{
  int m = blockIdx.x * blockDim.x + threadIdx.x;
  if (m >= E + n) return;
  int c0 = blockIdx.y * 4;
  int s, d;
  if (m < E) { s = ei[m]; d = ei[E + m]; } else { s = d = m - E; }
  float w = nrm[m];
  const float* hp = H + (size_t)s * LDF + c0;
  float* op = out + (size_t)d * LDF + c0;
  if (c0 + 3 < dout) {
    float4 v = *(const float4*)hp;
    atomicAdd(op + 0, v.x * w);
    atomicAdd(op + 1, v.y * w);
    atomicAdd(op + 2, v.z * w);
    atomicAdd(op + 3, v.w * w);
  } else {
    for (int j = 0; j < 4 && c0 + j < dout; ++j)
      atomicAdd(op + j, hp[j] * w);
  }
}

// bias + tanh, writing f16 activation buffer [n, Sout] (zero-padded cols)
__global__ void act_kernel(const float* __restrict__ agg, const float* __restrict__ b,
                           _Float16* __restrict__ out, int n, int dout, int Sout) {
  int idx = blockIdx.x * blockDim.x + threadIdx.x;
  if (idx >= n * Sout) return;
  int r = idx / Sout, c = idx - r * Sout;
  float v = 0.0f;
  if (c < dout) v = tanhf(agg[(size_t)r * LDF + c] + b[c]);
  out[idx] = (_Float16)v;
}

// final layer: bias only, f32 to d_out [n,3]
__global__ void bias_out_kernel(const float* __restrict__ agg, const float* __restrict__ b,
                                float* __restrict__ out, int n, int dout) {
  int idx = blockIdx.x * blockDim.x + threadIdx.x;
  if (idx >= n * dout) return;
  int r = idx / dout, c = idx - r * dout;
  out[idx] = agg[(size_t)r * LDF + c] + b[c];
}

// concat([h6 (f16, stride 512), x (f32)]) -> f16 [n, 544] zero-padded
__global__ void concat_kernel(const _Float16* __restrict__ h, const float* __restrict__ x,
                              _Float16* __restrict__ out, int n) {
  int idx = blockIdx.x * blockDim.x + threadIdx.x;
  if (idx >= n * LDC) return;
  int r = idx / LDC, c = idx - r * LDC;
  _Float16 v = (_Float16)0.0f;
  if (c < 512)      v = h[(size_t)r * 512 + c];
  else if (c < 520) v = (_Float16)x[r * 8 + (c - 512)];
  out[idx] = v;
}

// ---------------------------------------------------------------------------
extern "C" void kernel_launch(void* const* d_in, const int* in_sizes, int n_in,
                              void* d_out, int out_size, void* d_ws, size_t ws_size,
                              hipStream_t stream) {
  const float* x  = (const float*)d_in[0];
  const int*   ei = (const int*)d_in[1];   // [2,E]: row0=src, row1=dst
  const int n = in_sizes[0] / 8;
  const int E = in_sizes[1] / 2;
  const float* Ws[12];
  const float* Bs[12];
  for (int i = 0; i < 12; ++i) { Ws[i] = (const float*)d_in[3 + i]; Bs[i] = (const float*)d_in[15 + i]; }

  static const int DIN [12] = {8, 16, 32, 64, 128, 256, 520, 256, 128, 64, 32, 16};
  static const int DOUT[12] = {16, 32, 64, 128, 256, 512, 256, 128, 64, 32, 16, 3};
  int Kp[12], Dp[12];
  for (int i = 0; i < 12; ++i) {
    Kp[i] = (DIN[i]  + 31) & ~31;   // padded K   (multiple of 32)
    Dp[i] = (DOUT[i] + 31) & ~31;   // padded out (multiple of 32)
  }

  const int np = (n + 15) & ~15;    // rows padded to 16 (n=30000 already is)

  // ---- carve workspace
  char* w = (char*)d_ws;
  auto carve = [&](size_t bytes) -> void* {
    void* p = (void*)w; w += (bytes + 255) & ~(size_t)255; return p;
  };
  float*     deg = (float*)carve((size_t)n * 4);
  float*     nrm = (float*)carve((size_t)(E + n) * 4);
  float*     G   = (float*)carve((size_t)np * LDF * 4);     // gemm output
  float*     AGG = (float*)carve((size_t)np * LDF * 4);     // aggregation
  _Float16*  F[2];
  F[0] = (_Float16*)carve((size_t)np * LDC * 2);            // f16 activations
  F[1] = (_Float16*)carve((size_t)np * LDC * 2);
  _Float16*  Xh  = (_Float16*)carve((size_t)np * 32 * 2);   // x in f16 [n,32]
  _Float16*  WT[12];
  for (int i = 0; i < 12; ++i) WT[i] = (_Float16*)carve((size_t)Dp[i] * Kp[i] * 2);

  // ---- normalization: deg (with self loop), rsqrt, per-message norm
  fill_kernel<<<(n + 255) / 256, 256, 0, stream>>>(deg, 1.0f, n);
  deg_kernel<<<(E + 255) / 256, 256, 0, stream>>>(ei + E, deg, E);
  rsqrt_kernel<<<(n + 255) / 256, 256, 0, stream>>>(deg, n);
  norm_kernel<<<(E + n + 255) / 256, 256, 0, stream>>>(ei, deg, nrm, E, n);

  // ---- one-time operand conversions
  for (int i = 0; i < 12; ++i) {
    int tot = Dp[i] * Kp[i];
    wt_kernel<<<(tot + 255) / 256, 256, 0, stream>>>(Ws[i], WT[i], DIN[i], DOUT[i], Kp[i], tot);
  }
  xh_kernel<<<(n * 32 + 255) / 256, 256, 0, stream>>>(x, Xh, n);

  // ---- layer pipeline
  const _Float16* in = Xh;
  int p = 0;                 // next act output goes to F[p]
  const int nRowTiles = np / 16;

  for (int i = 0; i < 12; ++i) {
    int dout = DOUT[i];

    dim3 gg((nRowTiles + 7) / 8, Dp[i] / 32);
    gemm_wmma_kernel<<<gg, 256, 0, stream>>>(in, WT[i], G, n, nRowTiles, Kp[i], Dp[i]);

    int fillN = n * LDF;
    fill_kernel<<<(fillN + 255) / 256, 256, 0, stream>>>(AGG, 0.0f, fillN);

    dim3 sg((E + n + 255) / 256, (dout + 3) / 4);
    scatter_kernel<<<sg, 256, 0, stream>>>(G, ei, nrm, AGG, E, n, dout);

    if (i < 11) {
      int Sout = (i == 5) ? 512 : Kp[i + 1];       // stride of next gemm input
      int tot = n * Sout;
      act_kernel<<<(tot + 255) / 256, 256, 0, stream>>>(AGG, Bs[i], F[p], n, dout, Sout);
      in = F[p]; p ^= 1;
      if (i == 5) {                                 // build concat([h6, x]) f16
        int tot2 = n * LDC;
        concat_kernel<<<(tot2 + 255) / 256, 256, 0, stream>>>(F[p ^ 1], x, F[p], n);
        in = F[p]; p ^= 1;
      }
    } else {
      int tot = n * 3;
      bias_out_kernel<<<(tot + 255) / 256, 256, 0, stream>>>(AGG, Bs[i], (float*)d_out, n, 3);
    }
  }
}
